// HF_guided_CA_2791728742761
// MI455X (gfx1250) — matrix operand order repaired and checked by reading, not compiled, not run on hardware
//
#include <hip/hip_runtime.h>
#include <hip/hip_bf16.h>
#include <math.h>

#define B_   8
#define C_   256
#define HW_  4096
#define G_   32
#define EPS_ 1e-5f
// softmax scale 1/sqrt(256) folded into Q together with log2(e) so we can use exp2
#define QSCALE (0.0625f * 1.44269504088896f)

#define KP_ 264   // LDS pitch (bf16 elems) for K tile rows (256 + 4-DWORD pad)
#define VP_ 72    // LDS pitch (bf16 elems) for V tile rows (64 + 4-DWORD pad)

typedef __bf16 bf16;
typedef __attribute__((ext_vector_type(8)))  __bf16 v8bf;
typedef __attribute__((ext_vector_type(16))) __bf16 v16bf;
typedef __attribute__((ext_vector_type(8)))  float  v8f;
typedef __attribute__((ext_vector_type(4)))  unsigned int v4u;
typedef __attribute__((ext_vector_type(8)))  int v8i;
typedef __attribute__((ext_vector_type(4)))  int v4i;

#if defined(__HIP_DEVICE_COMPILE__) && __has_builtin(__builtin_amdgcn_tensor_load_to_lds)
#define USE_TDM 1
#else
#define USE_TDM 0
#endif

// ---------------- WMMA helpers (CDNA5 bf16 16x16x32, fp32 accum) ----------------

__device__ __forceinline__ v8f wmma_bf16(v16bf a, v16bf b, v8f c) {
  return __builtin_amdgcn_wmma_f32_16x16x32_bf16(false, a, false, b, (short)0, c,
                                                 false, false);
}

__device__ __forceinline__ v16bf combine16(v8bf lo, v8bf hi) {
  v16bf r;
#pragma unroll
  for (int i = 0; i < 8; ++i) { r[i] = lo[i]; r[i + 8] = hi[i]; }
  return r;
}

// A operand: 16x32 tile, source row-major [m][k] with `pitch` elements per row.
__device__ __forceinline__ v16bf load_a_tile(const bf16* base, int pitch, int lane) {
  int m    = lane & 15;
  int koff = (lane >> 4) * 8;
  const bf16* p = base + (size_t)m * pitch + koff;
  v8bf lo = *(const v8bf*)(p);
  v8bf hi = *(const v8bf*)(p + 16);
  return combine16(lo, hi);
}

// B operand: 32x16 tile of B[k][n], sourced from row-major src[n][k] (contiguous k).
__device__ __forceinline__ v16bf load_b_tile(const bf16* base, int pitch, int lane) {
  int n    = lane & 15;
  int koff = (lane >> 4) * 8;
  const bf16* p = base + (size_t)n * pitch + koff;
  v8bf lo = *(const v8bf*)(p);
  v8bf hi = *(const v8bf*)(p + 16);
  return combine16(lo, hi);
}

#if USE_TDM
// ---------------- Tensor Data Mover: 2D tile (bf16) -> LDS, with row padding ------
// Packs D# groups per CDNA5 ISA and issues TENSOR_LOAD_TO_LDS.
__device__ __forceinline__ void tdm_load_2d_bf16(
    const bf16* gsrc, unsigned lds_off,
    unsigned tensor_d0, unsigned long long stride_d0, unsigned tensor_d1,
    unsigned tile_d0, unsigned tile_d1,
    unsigned pad_interval_code, unsigned pad_amount_code) {
  unsigned long long ga = (unsigned long long)(size_t)gsrc;
  v4u g0;
  g0[0] = 1u;                                   // count=1 (valid), user mode
  g0[1] = lds_off;                              // LDS byte address
  g0[2] = (unsigned)ga;                         // global_addr[31:0]
  g0[3] = (unsigned)(ga >> 32) | (2u << 30);    // global_addr[56:32] | type=2
  v8i g1;
  g1[0] = (int)((0u) |                          // workgroup_mask = 0 (no cluster)
                (1u << 16) |                    // data_size = 2 bytes
                (1u << 20) |                    // pad_enable
                (pad_interval_code << 22) |
                (pad_amount_code << 25));
  g1[1] = (int)((tensor_d0 & 0xFFFFu) << 16);   // [47:32]=atomic addr(0); dim0[15:0]
  g1[2] = (int)((tensor_d0 >> 16) | ((tensor_d1 & 0xFFFFu) << 16));
  g1[3] = (int)((tensor_d1 >> 16) | (tile_d0 << 16));
  g1[4] = (int)(tile_d1 & 0xFFFFu);             // tile_dim1 | tile_dim2=0
  g1[5] = (int)(unsigned)stride_d0;             // dim0_stride[31:0]
  g1[6] = (int)(unsigned)((stride_d0 >> 32) & 0xFFFFu); // dim0_stride[47:32]
  g1[7] = 0;                                    // dim1_stride (unused, 2D)
  v4i z = {0, 0, 0, 0};
#if __clang_major__ >= 23
  v8i z8 = {0, 0, 0, 0, 0, 0, 0, 0};
  __builtin_amdgcn_tensor_load_to_lds(g0, g1, z, z, z8, 0);
#else
  __builtin_amdgcn_tensor_load_to_lds(g0, g1, z, z, 0);
#endif
}
#endif

// ---------------- 1. GroupNorm statistics ----------------

__global__ void gn_stats_kernel(const float* __restrict__ x,
                                float* __restrict__ mean_o,
                                float* __restrict__ rstd_o) {
  const int bg = blockIdx.x;            // b*G + g
  const int b  = bg / G_, g = bg % G_;
  const int CPG = C_ / G_;              // 8
  const float* base = x + ((size_t)b * C_ + (size_t)g * CPG) * HW_;
  const int N = CPG * HW_;              // 32768
  float s = 0.f, s2 = 0.f;
  for (int i = threadIdx.x; i < N; i += blockDim.x) {
    float v = base[i];
    s += v; s2 += v * v;
  }
  __shared__ float sh0[256], sh1[256];
  sh0[threadIdx.x] = s; sh1[threadIdx.x] = s2;
  __syncthreads();
  for (int off = 128; off > 0; off >>= 1) {
    if (threadIdx.x < off) {
      sh0[threadIdx.x] += sh0[threadIdx.x + off];
      sh1[threadIdx.x] += sh1[threadIdx.x + off];
    }
    __syncthreads();
  }
  if (threadIdx.x == 0) {
    float m   = sh0[0] / (float)N;
    float var = sh1[0] / (float)N - m * m;
    mean_o[bg] = m;
    rstd_o[bg] = rsqrtf(var + EPS_);
  }
}

// ---------------- 2. normalize + transpose -> normT[b][p][c] (bf16) ----------------

__global__ void norm_transpose_kernel(const float* __restrict__ x,
                                      const float* __restrict__ gamma,
                                      const float* __restrict__ beta,
                                      const float* __restrict__ mean,
                                      const float* __restrict__ rstd,
                                      bf16* __restrict__ normT) {
  __shared__ float tile[32][33];
  const int b  = blockIdx.z;
  const int c0 = blockIdx.y * 32;
  const int p0 = blockIdx.x * 32;
  const int tx = threadIdx.x, ty = threadIdx.y;   // 32 x 8
#pragma unroll
  for (int k = 0; k < 4; ++k) {
    int c = c0 + ty + k * 8;
    int g = c >> 3;                                // C/G == 8
    float m = mean[b * G_ + g], r = rstd[b * G_ + g];
    float v = x[((size_t)b * C_ + c) * HW_ + p0 + tx];
    tile[ty + k * 8][tx] = (v - m) * r * gamma[c] + beta[c];
  }
  __syncthreads();
#pragma unroll
  for (int k = 0; k < 4; ++k) {
    int p = p0 + ty + k * 8;
    normT[((size_t)b * HW_ + p) * C_ + c0 + tx] = (bf16)tile[tx][ty + k * 8];
  }
}

// ---------------- 3. weight conversion to bf16 (layouts kept [o][c]) ----------------

__global__ void convert_weights_kernel(const float* __restrict__ w_kv,
                                       const float* __restrict__ w_out,
                                       bf16* __restrict__ w_kv_bf,
                                       bf16* __restrict__ w_out_bf) {
  int idx = blockIdx.x * blockDim.x + threadIdx.x;
  const int NKV = 512 * 256;
  if (idx < NKV)            w_kv_bf[idx]        = (bf16)w_kv[idx];
  else if (idx < NKV + 256 * 256) w_out_bf[idx - NKV] = (bf16)w_out[idx - NKV];
}

// ---------------- 4. Q projection (K=3) -> q[b][p][c] bf16, scale folded ----------------

__global__ void qproj_kernel(const float* __restrict__ quary,
                             const float* __restrict__ w_q,
                             bf16* __restrict__ q_bf) {
  size_t idx = (size_t)blockIdx.x * blockDim.x + threadIdx.x; // B*HW*C
  int c = (int)(idx & 255);
  size_t bp = idx >> 8;
  int p = (int)(bp & 4095);
  int b = (int)(bp >> 12);
  float acc = 0.f;
#pragma unroll
  for (int ic = 0; ic < 3; ++ic)
    acc += quary[((size_t)b * 3 + ic) * HW_ + p] * w_q[c * 3 + ic];
  q_bf[idx] = (bf16)(acc * QSCALE);
}

// ---------------- 5. KV GEMM (WMMA): M=p, N=o, K=c ----------------

__global__ void kv_gemm_kernel(const bf16* __restrict__ normT,
                               const bf16* __restrict__ w_kv_bf,
                               bf16* __restrict__ k_bf,
                               bf16* __restrict__ vt_bf) {
  const int lane = threadIdx.x & 31;
  const int wave = threadIdx.x >> 5;
  const int b  = blockIdx.z;
  const int o0 = blockIdx.x * 64;                 // 8 o-blocks of 64
  const int p0 = (blockIdx.y * 4 + wave) * 16;    // 64 y-blocks * 4 waves

  const bf16* abase = normT + ((size_t)b * HW_ + p0) * C_;
  v16bf aa[8];
#pragma unroll
  for (int ch = 0; ch < 8; ++ch) aa[ch] = load_a_tile(abase + ch * 32, C_, lane);

  const int hh = lane >> 4, nn = lane & 15;
#pragma unroll
  for (int t = 0; t < 4; ++t) {
    v8f acc = {0, 0, 0, 0, 0, 0, 0, 0};
#pragma unroll
    for (int ch = 0; ch < 8; ++ch) {
      v16bf bb = load_b_tile(w_kv_bf + (size_t)(o0 + t * 16) * C_ + ch * 32, C_, lane);
      acc = wmma_bf16(aa[ch], bb, acc);
    }
    int o = o0 + t * 16 + nn;
    if (o0 < 256) {            // K output: [b][p][o], strided b16 stores
#pragma unroll
      for (int r = 0; r < 8; ++r) {
        int p = p0 + hh * 8 + r;
        k_bf[((size_t)b * HW_ + p) * C_ + o] = (bf16)acc[r];
      }
    } else {                   // V output transposed: [b][o'][p], one b128 store
      v8bf pk;
#pragma unroll
      for (int r = 0; r < 8; ++r) pk[r] = (bf16)acc[r];
      *(v8bf*)(vt_bf + ((size_t)b * C_ + (o - 256)) * HW_ + p0 + hh * 8) = pk;
    }
  }
}

// ---------------- 6. Flash attention (WMMA + TDM double-buffered K/V tiles) -------
// WG = 4 waves; each wave owns 16 query rows; all share LDS-staged 64-key tiles.

__global__ void __launch_bounds__(128)
flash_attn_kernel(const bf16* __restrict__ q_bf,
                  const bf16* __restrict__ k_bf,
                  const bf16* __restrict__ vt_bf,
                  bf16* __restrict__ attn_out) {
  const int lane = threadIdx.x & 31;
  const int wave = threadIdx.x >> 5;
  const int b  = blockIdx.y;
  const int p0 = (blockIdx.x * 4 + wave) * 16;
  const int hh = lane >> 4, nn = lane & 15;

  // LDS: double-buffered K tile (64 x 256) and V tile (256 x 64), padded pitches.
  __shared__ __attribute__((aligned(16))) bf16 kst[2][64][KP_];
  __shared__ __attribute__((aligned(16))) bf16 vst[2][256][VP_];
  __shared__ __attribute__((aligned(16))) bf16 psh[4][16][72];

  // Q A-tiles (16 rows x 256 k, 8 chunks)
  const bf16* qbase = q_bf + ((size_t)b * HW_ + p0) * C_;
  v16bf qa[8];
#pragma unroll
  for (int ch = 0; ch < 8; ++ch) qa[ch] = load_a_tile(qbase + ch * 32, C_, lane);

  v8f O[16];
#pragma unroll
  for (int nt = 0; nt < 16; ++nt) O[nt] = (v8f){0, 0, 0, 0, 0, 0, 0, 0};
  float m[8], l[8];
#pragma unroll
  for (int r = 0; r < 8; ++r) { m[r] = -INFINITY; l[r] = 0.f; }

  const bf16* kgl0 = k_bf  + (size_t)b * HW_ * C_;
  const bf16* vgl0 = vt_bf + (size_t)b * C_ * HW_;

#if USE_TDM
  if (wave == 0) {   // prologue: DMA first tiles into buffer 0
    tdm_load_2d_bf16(kgl0, (unsigned)(size_t)&kst[0][0][0],
                     C_, C_, HW_, C_, 64, /*pad: 4dw per 128dw*/ 6, 3);
    tdm_load_2d_bf16(vgl0, (unsigned)(size_t)&vst[0][0][0],
                     HW_, HW_, C_, 64, C_, /*pad: 4dw per 32dw*/ 4, 3);
  }
#endif

  for (int it = 0; it < HW_ / 64; ++it) {
    const int j0 = it * 64;
#if USE_TDM
    if (wave == 0) {
      if (it + 1 < HW_ / 64) {   // DMA next tiles into other buffer, then release cur
        const int nb = (it + 1) & 1;
        tdm_load_2d_bf16(kgl0 + (size_t)(j0 + 64) * C_,
                         (unsigned)(size_t)&kst[nb][0][0],
                         C_, C_, HW_, C_, 64, 6, 3);
        tdm_load_2d_bf16(vgl0 + (j0 + 64),
                         (unsigned)(size_t)&vst[nb][0][0],
                         HW_, HW_, C_, 64, C_, 4, 3);
        __builtin_amdgcn_s_wait_tensorcnt(2);
      } else {
        __builtin_amdgcn_s_wait_tensorcnt(0);
      }
    }
    __syncthreads();             // staged tiles visible to all waves
    const int buf = it & 1;
#else
    {                            // fallback: cooperative synchronous staging
      const bf16* kg = kgl0 + (size_t)j0 * C_;
      for (int idx = threadIdx.x; idx < 64 * 32; idx += 128) {
        int r = idx >> 5, cc = (idx & 31) * 8;
        *(v8bf*)&kst[0][r][cc] = *(const v8bf*)(kg + (size_t)r * C_ + cc);
      }
      const bf16* vg = vgl0 + j0;
      for (int idx = threadIdx.x; idx < 256 * 8; idx += 128) {
        int r = idx >> 3, jj = (idx & 7) * 8;
        *(v8bf*)&vst[0][r][jj] = *(const v8bf*)(vg + (size_t)r * HW_ + jj);
      }
    }
    __syncthreads();
    const int buf = 0;
#endif

    // S = Q * K^T for 4 tiles of 16 keys (B operands from LDS)
    v8f S[4];
#pragma unroll
    for (int t = 0; t < 4; ++t) {
      v8f s = {0, 0, 0, 0, 0, 0, 0, 0};
#pragma unroll
      for (int ch = 0; ch < 8; ++ch)
        s = wmma_bf16(qa[ch], load_b_tile(&kst[buf][t * 16][ch * 32], KP_, lane), s);
      S[t] = s;
    }
    // online softmax update (per row r+8*hh; reduce over 16 lanes within half)
#pragma unroll
    for (int r = 0; r < 8; ++r) {
      float mx = fmaxf(fmaxf(S[0][r], S[1][r]), fmaxf(S[2][r], S[3][r]));
#pragma unroll
      for (int d = 8; d >= 1; d >>= 1) mx = fmaxf(mx, __shfl_xor(mx, d, 32));
      float mnew = fmaxf(m[r], mx);
      float corr = exp2f(m[r] - mnew);
      m[r] = mnew;
      float rs = 0.f;
#pragma unroll
      for (int t = 0; t < 4; ++t) {
        float e = exp2f(S[t][r] - mnew);
        S[t][r] = e;
        rs += e;
      }
#pragma unroll
      for (int d = 8; d >= 1; d >>= 1) rs += __shfl_xor(rs, d, 32);
      l[r] = l[r] * corr + rs;
#pragma unroll
      for (int nt = 0; nt < 16; ++nt) O[nt][r] *= corr;
    }
    // P: D-layout -> LDS -> A-layout
#pragma unroll
    for (int t = 0; t < 4; ++t)
#pragma unroll
      for (int r = 0; r < 8; ++r)
        psh[wave][r + 8 * hh][t * 16 + nn] = (bf16)S[t][r];

    v16bf pa[2];
#pragma unroll
    for (int a = 0; a < 2; ++a) {
      const bf16* row = &psh[wave][nn][0] + a * 32 + hh * 8;
      v8bf lo = *(const v8bf*)(row);
      v8bf hi = *(const v8bf*)(row + 16);
      pa[a] = combine16(lo, hi);
    }
    // O += P * V   (B operands from LDS-staged Vt, contiguous in j)
#pragma unroll
    for (int nt = 0; nt < 16; ++nt) {
#pragma unroll
      for (int a = 0; a < 2; ++a)
        O[nt] = wmma_bf16(pa[a],
                          load_b_tile(&vst[buf][nt * 16][a * 32], VP_, lane), O[nt]);
    }
    __syncthreads();             // all waves done with buf before DMA may overwrite
  }

  // epilogue: O / l -> attn_out[b][p][c] bf16
  float linv[8];
#pragma unroll
  for (int r = 0; r < 8; ++r) linv[r] = 1.0f / l[r];
  bf16* obase = attn_out + ((size_t)b * HW_ + p0) * C_;
#pragma unroll
  for (int nt = 0; nt < 16; ++nt) {
    int c = nt * 16 + nn;
#pragma unroll
    for (int r = 0; r < 8; ++r)
      obase[(size_t)(8 * hh + r) * C_ + c] = (bf16)(O[nt][r] * linv[r]);
  }
}

// ---------------- 7. out projection + bias + residual ----------------

__global__ void __launch_bounds__(128)
out_proj_kernel(const bf16* __restrict__ attn_out,
                const bf16* __restrict__ w_out_bf,
                const float* __restrict__ b_out,
                const float* __restrict__ input,
                float* __restrict__ out) {
  const int lane = threadIdx.x & 31;
  const int wave = threadIdx.x >> 5;
  const int b  = blockIdx.y;
  const int p0 = (blockIdx.x * 4 + wave) * 16;
  const int hh = lane >> 4, nn = lane & 15;

  const bf16* abase = attn_out + ((size_t)b * HW_ + p0) * C_;
  v16bf aa[8];
#pragma unroll
  for (int ch = 0; ch < 8; ++ch) aa[ch] = load_a_tile(abase + ch * 32, C_, lane);

#pragma unroll
  for (int nt = 0; nt < 16; ++nt) {
    v8f acc = {0, 0, 0, 0, 0, 0, 0, 0};
#pragma unroll
    for (int ch = 0; ch < 8; ++ch) {
      v16bf bb = load_b_tile(w_out_bf + (size_t)(nt * 16) * C_ + ch * 32, C_, lane);
      acc = wmma_bf16(aa[ch], bb, acc);
    }
    int co = nt * 16 + nn;                 // fixed channel per lane
    float bias = b_out[co];
    size_t base = ((size_t)b * C_ + co) * HW_ + p0 + 8 * hh;  // 8 contiguous p
#pragma unroll
    for (int r = 0; r < 8; ++r)
      out[base + r] = acc[r] + bias + input[base + r];
  }
}

// ---------------- launcher ----------------

extern "C" void kernel_launch(void* const* d_in, const int* in_sizes, int n_in,
                              void* d_out, int out_size, void* d_ws, size_t ws_size,
                              hipStream_t stream) {
  (void)in_sizes; (void)n_in; (void)out_size; (void)ws_size;
  const float* input  = (const float*)d_in[0];
  const float* quary  = (const float*)d_in[1];
  const float* gammaw = (const float*)d_in[2];
  const float* betaw  = (const float*)d_in[3];
  const float* w_q    = (const float*)d_in[4];
  const float* w_kv   = (const float*)d_in[5];
  const float* w_out  = (const float*)d_in[6];
  const float* b_out  = (const float*)d_in[7];
  float* out = (float*)d_out;

  const size_t NB = (size_t)B_ * HW_ * C_ * sizeof(bf16);  // 16 MB per bf16 tensor
  char* ws = (char*)d_ws;
  float* mean   = (float*)(ws);
  float* rstd   = (float*)(ws + 1024);
  bf16* normT   = (bf16*)(ws + 2048);            // reused as attn_out later
  bf16* q_bf    = (bf16*)(ws + 2048 + NB);
  bf16* k_bf    = (bf16*)(ws + 2048 + 2 * NB);
  bf16* vt_bf   = (bf16*)(ws + 2048 + 3 * NB);
  bf16* wkv_bf  = (bf16*)(ws + 2048 + 4 * NB);
  bf16* wout_bf = (bf16*)(ws + 2048 + 4 * NB + 512 * 256 * sizeof(bf16));
  bf16* attn_o  = normT;   // normT dead after kv_gemm

  convert_weights_kernel<<<(512 * 256 + 256 * 256 + 255) / 256, 256, 0, stream>>>(
      w_kv, w_out, wkv_bf, wout_bf);
  gn_stats_kernel<<<B_ * G_, 256, 0, stream>>>(input, mean, rstd);
  qproj_kernel<<<(unsigned)(((size_t)B_ * HW_ * C_) / 256), 256, 0, stream>>>(
      quary, w_q, q_bf);
  norm_transpose_kernel<<<dim3(HW_ / 32, C_ / 32, B_), dim3(32, 8), 0, stream>>>(
      input, gammaw, betaw, mean, rstd, normT);
  kv_gemm_kernel<<<dim3(8, HW_ / 64, B_), 128, 0, stream>>>(normT, wkv_bf, k_bf, vt_bf);
  flash_attn_kernel<<<dim3(HW_ / 64, B_), 128, 0, stream>>>(q_bf, k_bf, vt_bf, attn_o);
  out_proj_kernel<<<dim3(HW_ / 64, B_), 128, 0, stream>>>(attn_o, wout_bf, b_out,
                                                          input, out);
}
